// WindowAttention_71863392797278
// MI455X (gfx1250) — matrix operand order repaired and checked by reading, not compiled, not run on hardware
//
#include <hip/hip_runtime.h>
#include <stdint.h>

typedef __attribute__((ext_vector_type(16))) __bf16 bf16x16;
typedef __attribute__((ext_vector_type(8)))  float  floatx8;
typedef __attribute__((ext_vector_type(4)))  int    i32x4;

union AFrag { bf16x16 v; unsigned int u[8]; };

__device__ __forceinline__ unsigned short f2bf(float f) {
    union { float f; unsigned int u; } v; v.f = f;
    unsigned int r = v.u + 0x7FFFu + ((v.u >> 16) & 1u);   // round-to-nearest-even
    return (unsigned short)(r >> 16);
}

// ---- async global->LDS copy support (CDNA5 GLOBAL_LOAD_ASYNC_TO_LDS_B128) ----
#if defined(__has_builtin)
#  if __has_builtin(__builtin_amdgcn_global_load_async_to_lds_b128) && \
      __has_builtin(__builtin_amdgcn_s_wait_asynccnt)
#    define HAVE_ASYNC_LDS 1
#  endif
#endif
#ifndef HAVE_ASYNC_LDS
#  define HAVE_ASYNC_LDS 0
#endif

#if HAVE_ASYNC_LDS
__device__ __forceinline__ void async_copy16(const void* g, void* l) {
    __builtin_amdgcn_global_load_async_to_lds_b128(
        (__attribute__((address_space(1))) i32x4*)(void*)g,
        (__attribute__((address_space(3))) i32x4*)l, 0, 0);
}
#define ASYNC_WAIT(n) __builtin_amdgcn_s_wait_asynccnt(n)
#else
__device__ __forceinline__ void async_copy16(const void* g, void* l) {
    const uint4* gp = (const uint4*)g;
    *(uint4*)l = *gp;
}
#define ASYNC_WAIT(n) ((void)0)
#endif

// ---------------------------------------------------------------------------
// Kernel 1: QKV projection.  C[100352 x 1536] = x[100352 x 512] * w + b
// Output scattered as bf16 into workspace laid out [b][h][{q,k,v}][l(49)][d(32)]
// ---------------------------------------------------------------------------
__global__ __launch_bounds__(128) void qkv_gemm_kernel(
    const float* __restrict__ x, const float* __restrict__ w,
    const float* __restrict__ bias, unsigned short* __restrict__ qkv)
{
    __shared__ alignas(16) unsigned short As[64 * 32];
    __shared__ alignas(16) unsigned short Bs[64 * 32];   // stored [n][k]
    const int tid = threadIdx.x;
    const int lane = tid & 31, wid = tid >> 5;
    const int l16 = lane & 15;
    const int h8  = (lane >> 4) * 8;
    const int h16 = (lane >> 4) * 16;
    const int mBlock = blockIdx.x, nBlock = blockIdx.y;
    const int N = 1536;

    floatx8 acc[4] = {};

    for (int k0 = 0; k0 < 512; k0 += 32) {
        __syncthreads();
        for (int t = tid; t < 2048; t += 128) {
            int i = t >> 5, j = t & 31;
            As[t] = f2bf(x[(mBlock * 64 + i) * 512 + k0 + j]);
            Bs[t] = f2bf(w[(k0 + j) * N + nBlock * 64 + i]);
        }
        // prefetch next K-step into caches (global_prefetch_b8)
        if (k0 + 32 < 512) {
            if (tid < 64)
                __builtin_prefetch(&x[(mBlock * 64 + tid) * 512 + k0 + 32], 0, 0);
            else if (tid < 96)
                __builtin_prefetch(&w[(k0 + 32 + (tid - 64)) * N + nBlock * 64], 0, 0);
        }
        __syncthreads();
        AFrag a;
        const unsigned short* arow = &As[(wid * 16 + l16) * 32];
        #pragma unroll
        for (int v = 0; v < 8; ++v) {
            int k = 2 * (v & 3) + ((v & 4) ? 16 : 0) + h8;
            a.u[v] = *(const unsigned int*)&arow[k];
        }
        #pragma unroll
        for (int s = 0; s < 4; ++s) {
            AFrag b;
            const unsigned short* brow = &Bs[(s * 16 + l16) * 32];
            #pragma unroll
            for (int v = 0; v < 8; ++v)
                b.u[v] = *(const unsigned int*)&brow[2 * (v & 3) + ((v & 4) ? 16 : 0) + h8];
            acc[s] = __builtin_amdgcn_wmma_f32_16x16x32_bf16(
                false, a.v, false, b.v, (short)0, acc[s], false, false);
        }
    }

    #pragma unroll
    for (int s = 0; s < 4; ++s) {
        int n = nBlock * 64 + s * 16 + l16;
        float bv = bias[n];
        int part = n >> 9;            // 0=q,1=k,2=v
        int h    = (n >> 5) & 15;     // head
        int d    = n & 31;            // dim within head
        #pragma unroll
        for (int r = 0; r < 8; ++r) {
            int m = mBlock * 64 + wid * 16 + r + h8;
            int b = m / 49, l = m % 49;
            size_t addr = ((((size_t)b * 16 + h) * 3 + part) * 49 + l) * 32 + d;
            qkv[addr] = f2bf(acc[s][r] + bv);
        }
    }
}

// ---------------------------------------------------------------------------
// Kernel 2: fused window attention per (b,h).  scores = qk^T*scale+bias+mask,
// softmax in registers, out = p@v.  One 128-thread block per (b,h); each wave
// owns a 16-row block of the (padded) 64x64 score matrix.
// ---------------------------------------------------------------------------
__global__ __launch_bounds__(128) void attn_kernel(
    const unsigned short* __restrict__ qkv, const float* __restrict__ bias_table,
    const float* __restrict__ mask, const int* __restrict__ rel_index,
    unsigned short* __restrict__ attn_out)
{
    __shared__ alignas(16) unsigned short qs[64 * 32];   // [l][d]
    __shared__ alignas(16) unsigned short ks[64 * 32];   // [c][d]
    __shared__ alignas(16) unsigned short vst[32 * 64];  // [d][c]  (transposed)
    __shared__ alignas(16) unsigned short ps[64 * 64];   // [m][c]  probabilities

    const int tid = threadIdx.x, lane = tid & 31, ti = tid >> 5;
    const int l16 = lane & 15;
    const int h8  = (lane >> 4) * 8;
    const int h16 = (lane >> 4) * 16;
    const int bid = blockIdx.x;
    const int b = bid >> 4, h = bid & 15, wnd = bid & 63;

    const size_t base = (size_t)bid * 3 * 1568;
    for (int t = tid; t < 2048; t += 128) { qs[t] = 0; ks[t] = 0; vst[t] = 0; }
    __syncthreads();

    // q and k copy straight through (async DMA, 16B chunks; 196 chunks each)
    for (int t = tid; t < 392; t += 128) {
        if (t < 196) async_copy16(&qkv[base + (size_t)t * 8], &qs[t * 8]);
        else         async_copy16(&qkv[base + 1568 + (size_t)(t - 196) * 8],
                                  &ks[(t - 196) * 8]);
    }
    // v needs a transpose -> element-wise staging
    for (int t = tid; t < 1568; t += 128) {
        int l = t >> 5, d = t & 31;
        vst[d * 64 + l] = qkv[base + 2 * 1568 + t];
    }
    ASYNC_WAIT(0);
    __syncthreads();

    // ---- scores: q(16x32) @ k^T -> 16x64 in 4 WMMA tiles ----
    floatx8 sc[4];
    AFrag qa;
    {
        const unsigned short* row = &qs[(ti * 16 + l16) * 32];
        #pragma unroll
        for (int v = 0; v < 8; ++v)
            qa.u[v] = *(const unsigned int*)&row[2 * (v & 3) + ((v & 4) ? 16 : 0) + h8];
    }
    #pragma unroll
    for (int tj = 0; tj < 4; ++tj) {
        AFrag kb;
        const unsigned short* row = &ks[(tj * 16 + l16) * 32];
        #pragma unroll
        for (int v = 0; v < 8; ++v)
            kb.u[v] = *(const unsigned int*)&row[2 * (v & 3) + ((v & 4) ? 16 : 0) + h8];
        floatx8 z = {};
        sc[tj] = __builtin_amdgcn_wmma_f32_16x16x32_bf16(
            false, qa.v, false, kb.v, (short)0, z, false, false);
    }

    // ---- fused scale + rel-pos bias + mask + softmax (per row) ----
    const float scale = 0.17677669529663687f;   // 1/sqrt(32)
    #pragma unroll
    for (int r = 0; r < 8; ++r) {
        int m = ti * 16 + r + h8;
        float xv[4];
        #pragma unroll
        for (int tj = 0; tj < 4; ++tj) {
            int c = tj * 16 + l16;
            float add = -1e30f;
            if (m < 49 && c < 49) {
                int ri = rel_index[m * 49 + c];
                add = bias_table[ri * 16 + h] + mask[(size_t)wnd * 2401 + m * 49 + c];
            }
            xv[tj] = sc[tj][r] * scale + add;
        }
        float mx = fmaxf(fmaxf(xv[0], xv[1]), fmaxf(xv[2], xv[3]));
        #pragma unroll
        for (int off = 8; off >= 1; off >>= 1) mx = fmaxf(mx, __shfl_xor(mx, off));
        float e[4], sum = 0.f;
        #pragma unroll
        for (int tj = 0; tj < 4; ++tj) { e[tj] = __expf(xv[tj] - mx); sum += e[tj]; }
        #pragma unroll
        for (int off = 8; off >= 1; off >>= 1) sum += __shfl_xor(sum, off);
        float inv = 1.0f / sum;
        #pragma unroll
        for (int tj = 0; tj < 4; ++tj)
            ps[m * 64 + tj * 16 + l16] = f2bf(e[tj] * inv);
    }
    __syncthreads();

    // ---- out = p(16x64) @ v(64x32): 2 K-steps x 2 N-tiles ----
    floatx8 oacc[2] = {};
    #pragma unroll
    for (int kst = 0; kst < 2; ++kst) {
        AFrag pa;
        const unsigned short* prow = &ps[(ti * 16 + l16) * 64 + kst * 32];
        #pragma unroll
        for (int v = 0; v < 8; ++v)
            pa.u[v] = *(const unsigned int*)&prow[2 * (v & 3) + ((v & 4) ? 16 : 0) + h8];
        #pragma unroll
        for (int nt = 0; nt < 2; ++nt) {
            AFrag vb;
            const unsigned short* vrow = &vst[(nt * 16 + l16) * 64 + kst * 32];
            #pragma unroll
            for (int v = 0; v < 8; ++v)
                vb.u[v] = *(const unsigned int*)&vrow[2 * (v & 3) + ((v & 4) ? 16 : 0) + h8];
            oacc[nt] = __builtin_amdgcn_wmma_f32_16x16x32_bf16(
                false, pa.v, false, vb.v, (short)0, oacc[nt], false, false);
        }
    }

    // store bf16 [b][l][h*32+d] for the out-projection GEMM
    #pragma unroll
    for (int nt = 0; nt < 2; ++nt) {
        int d = nt * 16 + l16;
        #pragma unroll
        for (int r = 0; r < 8; ++r) {
            int m = ti * 16 + r + h8;
            if (m < 49)
                attn_out[((size_t)b * 49 + m) * 512 + h * 32 + d] = f2bf(oacc[nt][r]);
        }
    }
}

// ---------------------------------------------------------------------------
// Kernel 3: output projection.  out[100352 x 512] = A_bf16 * w_out + b_out
// A tiles staged by async DMA with double buffering.
// ---------------------------------------------------------------------------
__global__ __launch_bounds__(128) void out_gemm_kernel(
    const unsigned short* __restrict__ A, const float* __restrict__ w,
    const float* __restrict__ bias, float* __restrict__ out)
{
    __shared__ alignas(16) unsigned short As[2][64 * 32];
    __shared__ alignas(16) unsigned short Bs[64 * 32];
    const int tid = threadIdx.x;
    const int lane = tid & 31, wid = tid >> 5;
    const int l16 = lane & 15;
    const int h8  = (lane >> 4) * 8;
    const int h16 = (lane >> 4) * 16;
    const int mBlock = blockIdx.x, nBlock = blockIdx.y;
    const int N = 512;

    floatx8 acc[4] = {};

    // issue async copy of the 64x32 bf16 A tile at column k0 into buffer `buf`
    auto issueA = [&](int buf, int k0) {
        for (int t = tid; t < 256; t += 128) {          // 2 chunks per thread
            int i = t >> 2, c = t & 3;                  // row, 16B chunk
            async_copy16(&A[(size_t)(mBlock * 64 + i) * 512 + k0 + c * 8],
                         &As[buf][i * 32 + c * 8]);
        }
    };

    issueA(0, 0);

    for (int k0 = 0; k0 < 512; k0 += 32) {
        const int buf = (k0 >> 5) & 1;
        __syncthreads();                                 // prev reads of Bs done
        for (int t = tid; t < 2048; t += 128) {          // B: fp32 -> bf16 [n][k]
            int i = t >> 5, j = t & 31;
            Bs[t] = f2bf(w[(k0 + j) * N + nBlock * 64 + i]);
        }
        if (k0 + 32 < 512) {
            issueA(buf ^ 1, k0 + 32);                    // overlap next A copy
            ASYNC_WAIT(2);                               // retire current buffer
        } else {
            ASYNC_WAIT(0);
        }
        __syncthreads();                                 // As[buf] + Bs visible

        AFrag a;
        const unsigned short* arow = &As[buf][(wid * 16 + l16) * 32];
        #pragma unroll
        for (int v = 0; v < 8; ++v) {
            int k = 2 * (v & 3) + ((v & 4) ? 16 : 0) + h8;
            a.u[v] = *(const unsigned int*)&arow[k];
        }
        #pragma unroll
        for (int s = 0; s < 4; ++s) {
            AFrag b;
            const unsigned short* brow = &Bs[(s * 16 + l16) * 32];
            #pragma unroll
            for (int v = 0; v < 8; ++v)
                b.u[v] = *(const unsigned int*)&brow[2 * (v & 3) + ((v & 4) ? 16 : 0) + h8];
            acc[s] = __builtin_amdgcn_wmma_f32_16x16x32_bf16(
                false, a.v, false, b.v, (short)0, acc[s], false, false);
        }
    }

    #pragma unroll
    for (int s = 0; s < 4; ++s) {
        int n = nBlock * 64 + s * 16 + l16;
        float bv = bias[n];
        #pragma unroll
        for (int r = 0; r < 8; ++r) {
            int m = mBlock * 64 + wid * 16 + r + h8;
            out[(size_t)m * 512 + n] = acc[s][r] + bv;
        }
    }
}

// ---------------------------------------------------------------------------
extern "C" void kernel_launch(void* const* d_in, const int* in_sizes, int n_in,
                              void* d_out, int out_size, void* d_ws, size_t ws_size,
                              hipStream_t stream) {
    const float* x      = (const float*)d_in[0];
    const float* w_qkv  = (const float*)d_in[1];
    const float* b_qkv  = (const float*)d_in[2];
    const float* w_out  = (const float*)d_in[3];
    const float* b_out  = (const float*)d_in[4];
    const float* btab   = (const float*)d_in[5];
    const float* mask   = (const float*)d_in[6];
    const int*   relidx = (const int*)d_in[7];
    float* out = (float*)d_out;

    // workspace: qkv bf16 [2048][16][3][49][32], then attn-out bf16 [100352][512]
    unsigned short* qkv_ws  = (unsigned short*)d_ws;
    unsigned short* attn_ws = qkv_ws + (size_t)2048 * 16 * 3 * 1568;

    qkv_gemm_kernel<<<dim3(1568, 24), 128, 0, stream>>>(x, w_qkv, b_qkv, qkv_ws);
    attn_kernel<<<dim3(32768), 128, 0, stream>>>(qkv_ws, btab, mask, relidx, attn_ws);
    out_gemm_kernel<<<dim3(1568, 8), 128, 0, stream>>>(attn_ws, w_out, b_out, out);
}
